// GCN11_20693152432422
// MI455X (gfx1250) — matrix operand order repaired and checked by reading, not compiled, not run on hardware
//
#include <hip/hip_runtime.h>
#include <hip/hip_bf16.h>
#include <math.h>

typedef __attribute__((ext_vector_type(16))) __bf16 v16bf;
typedef __attribute__((ext_vector_type(8)))  float  v8f;

#define FDIM 128

// ---------------------------------------------------------------- utilities

__device__ __forceinline__ float waveReduceSum(float s) {
#pragma unroll
  for (int o = 16; o > 0; o >>= 1) s += __shfl_xor(s, o, 32);
  return s;
}

__device__ __forceinline__ void atomicMaxF32(float* addr, float val) {
  unsigned int* ua = (unsigned int*)addr;
  unsigned int old = __float_as_uint(*addr);
  while (__uint_as_float(old) < val) {
    unsigned int assumed = old;
    old = atomicCAS(ua, assumed, __float_as_uint(val));
    if (old == assumed) break;
  }
}

__device__ __forceinline__ v16bf pack_bf16(float4 q0, float4 q1, float4 q2, float4 q3) {
  v16bf a;
  a[0]  = (__bf16)q0.x; a[1]  = (__bf16)q0.y; a[2]  = (__bf16)q0.z; a[3]  = (__bf16)q0.w;
  a[4]  = (__bf16)q1.x; a[5]  = (__bf16)q1.y; a[6]  = (__bf16)q1.z; a[7]  = (__bf16)q1.w;
  a[8]  = (__bf16)q2.x; a[9]  = (__bf16)q2.y; a[10] = (__bf16)q2.z; a[11] = (__bf16)q2.w;
  a[12] = (__bf16)q3.x; a[13] = (__bf16)q3.y; a[14] = (__bf16)q3.z; a[15] = (__bf16)q3.w;
  return a;
}

__global__ void k_fill(float* __restrict__ p, float v, int n) {
  int i = blockIdx.x * blockDim.x + threadIdx.x;
  if (i < n) p[i] = v;
}

__global__ void k_fill_dev(float* __restrict__ p, const float* __restrict__ v, int n) {
  int i = blockIdx.x * blockDim.x + threadIdx.x;
  if (i < n) p[i] = v[0];
}

// ---------------------------------------------------------------- degree / norm

__global__ void k_deg(const int* __restrict__ dst, float* __restrict__ deg, int E) {
  int i = blockIdx.x * blockDim.x + threadIdx.x;
  if (i < E) atomicAdd(&deg[dst[i]], 1.0f);
}

__global__ void k_rsqrt(const float* __restrict__ deg, float* __restrict__ dis, int n) {
  int i = blockIdx.x * blockDim.x + threadIdx.x;
  if (i < n) dis[i] = rsqrtf(deg[i]);   // deg >= 1 (self-loop)
}

// ---------------------------------------------------------------- W pre-pack
// Pack W (f32, [K=128][col=128]) into bf16 in WMMA B-fragment order:
// Wpk[ct][kc][lane][i], ct=col-tile(8), kc=k-chunk(4), lane(32), i(16).
// Per CDNA5 ISA 7.12.2 16-bit 32x16 B layout: lane<16 -> col=ct*16+lane, K base kc*32;
// lane>=16 -> col=ct*16+lane-16, K base kc*32+8; element i -> K = kbase + i + (i>=8?8:0).
__global__ void k_pack_w(const float* __restrict__ W, __bf16* __restrict__ Wpk) {
  int idx = blockIdx.x * blockDim.x + threadIdx.x;   // 0..16383
  int i    = idx & 15;
  int lane = (idx >> 4) & 31;
  int kc   = (idx >> 9) & 3;
  int ct   = idx >> 11;
  int col   = ct * 16 + (lane & 15);
  int kbase = kc * 32 + ((lane < 16) ? 0 : 8);
  int K     = kbase + i + ((i >= 8) ? 8 : 0);
  Wpk[idx] = (__bf16)W[K * FDIM + col];
}

// ---------------------------------------------------------------- bf16 WMMA GEMM
// out[M x 128] = in[M x 128] @ W; B fragments pre-packed (global_load_b128, no LDS).
// Block = 256 threads = 8 waves; block covers 32 rows (2 M-tiles), wave w covers
// cols [16w,16w+16). M = 100000 = 3125 * 32 exactly.
__global__ __launch_bounds__(256) void k_gemm_wmma(const float* __restrict__ in,
                                                   const __bf16* __restrict__ Wpk,
                                                   float* __restrict__ out) {
  int wave = threadIdx.x >> 5;                 // col tile 0..7
  int lane = threadIdx.x & 31;
  int r0   = blockIdx.x * 32 + (lane & 15);    // row in M-tile 0; tile 1 = +16
  int col  = wave * 16 + (lane & 15);
  int khalf = (lane < 16) ? 0 : 8;
  const v16bf* wp = (const v16bf*)Wpk + wave * 4 * 32 + lane;

  v8f c0 = {}, c1 = {};
#pragma unroll
  for (int kc = 0; kc < 4; ++kc) {
    v16bf b = wp[kc * 32];                     // contiguous 32B fragment load
    int kbase = kc * 32 + khalf;               // multiple of 4 -> float4 aligned
    const float4* a0p = (const float4*)(in + (size_t)r0 * FDIM + kbase);
    const float4* a1p = (const float4*)(in + (size_t)(r0 + 16) * FDIM + kbase);
    v16bf a0 = pack_bf16(a0p[0], a0p[1], a0p[4], a0p[5]);   // K = kbase+0..7, +16..23
    v16bf a1 = pack_bf16(a1p[0], a1p[1], a1p[4], a1p[5]);
    c0 = __builtin_amdgcn_wmma_f32_16x16x32_bf16(false, a0, false, b, (short)0, c0,
                                                 false, false);
    c1 = __builtin_amdgcn_wmma_f32_16x16x32_bf16(false, a1, false, b, (short)0, c1,
                                                 false, false);
  }
  int orow = blockIdx.x * 32 + ((lane < 16) ? 0 : 8);
#pragma unroll
  for (int r = 0; r < 8; ++r) {
    out[(size_t)(orow + r) * FDIM + col]      = c0[r];
    out[(size_t)(orow + 16 + r) * FDIM + col] = c1[r];
  }
}

// ---------------------------------------------------------------- GCN aggregation
// one wave per edge: 32 lanes x float4 = 128 features, coalesced 512B per edge
__global__ __launch_bounds__(256) void k_gcn_scatter(const int* __restrict__ src,
                                                     const int* __restrict__ dst,
                                                     const float* __restrict__ dis,
                                                     const float* __restrict__ hw,
                                                     float* __restrict__ out, int E) {
  int e = blockIdx.x * 8 + (threadIdx.x >> 5);
  if (e >= E) return;
  int lane = threadIdx.x & 31;
  int s = src[e], d = dst[e];
  float coef = dis[s] * dis[d];
  float4 v = ((const float4*)(hw + (size_t)s * FDIM))[lane];
  float* od = out + (size_t)d * FDIM + lane * 4;
  atomicAdd(od + 0, v.x * coef);
  atomicAdd(od + 1, v.y * coef);
  atomicAdd(od + 2, v.z * coef);
  atomicAdd(od + 3, v.w * coef);
}

// self-loop contribution + bias + tanh
__global__ void k_gcn_finish(const float* __restrict__ hw, const float* __restrict__ dis,
                             const float* __restrict__ b, float* __restrict__ out, int N) {
  int idx = blockIdx.x * blockDim.x + threadIdx.x;     // N*32 threads, 4 feats each
  if (idx >= N * 32) return;
  int v = idx >> 5, f = (idx & 31) * 4;
  float c = dis[v] * dis[v];
  size_t base = (size_t)v * FDIM + f;
#pragma unroll
  for (int j = 0; j < 4; ++j)
    out[base + j] = tanhf(out[base + j] + hw[base + j] * c + b[f + j]);
}

// ---------------------------------------------------------------- AGNN

__global__ __launch_bounds__(256) void k_rownorm(const float* __restrict__ h,
                                                 float* __restrict__ nrm, int N) {
  int v = blockIdx.x * 8 + (threadIdx.x >> 5);
  if (v >= N) return;
  int lane = threadIdx.x & 31;
  float4 x = ((const float4*)(h + (size_t)v * FDIM))[lane];
  float s = waveReduceSum(x.x * x.x + x.y * x.y + x.z * x.z + x.w * x.w);
  if (lane == 0) nrm[v] = fmaxf(sqrtf(s), 1e-12f);
}

__global__ __launch_bounds__(256) void k_alpha(const int* __restrict__ src,
                                               const int* __restrict__ dst,
                                               const float* __restrict__ h,
                                               const float* __restrict__ nrm,
                                               const float* __restrict__ beta,
                                               float* __restrict__ alpha,
                                               float* __restrict__ amax, int E) {
  int e = blockIdx.x * 8 + (threadIdx.x >> 5);
  if (e >= E) return;
  int lane = threadIdx.x & 31;
  int s = src[e], d = dst[e];
  float4 a = ((const float4*)(h + (size_t)s * FDIM))[lane];
  float4 b = ((const float4*)(h + (size_t)d * FDIM))[lane];
  float dp = waveReduceSum(a.x * b.x + a.y * b.y + a.z * b.z + a.w * b.w);
  if (lane == 0) {
    float al = beta[0] * dp / (nrm[s] * nrm[d]);
    alpha[e] = al;
    atomicMaxF32(&amax[d], al);
  }
}

// denom starts with the self-loop term exp(beta - amax)
__global__ void k_denom_init(const float* __restrict__ beta, const float* __restrict__ amax,
                             float* __restrict__ denom, int N) {
  int i = blockIdx.x * blockDim.x + threadIdx.x;
  if (i < N) denom[i] = __expf(beta[0] - amax[i]);
}

__global__ void k_edge_exp(const int* __restrict__ dst, const float* __restrict__ alpha,
                           const float* __restrict__ amax, float* __restrict__ eexp,
                           float* __restrict__ denom, int E) {
  int e = blockIdx.x * blockDim.x + threadIdx.x;
  if (e >= E) return;
  int d = dst[e];
  float ev = __expf(alpha[e] - amax[d]);
  eexp[e] = ev;
  atomicAdd(&denom[d], ev);
}

__global__ __launch_bounds__(256) void k_agnn_scatter(const int* __restrict__ src,
                                                      const int* __restrict__ dst,
                                                      const float* __restrict__ eexp,
                                                      const float* __restrict__ denom,
                                                      const float* __restrict__ h,
                                                      float* __restrict__ out, int E) {
  int e = blockIdx.x * 8 + (threadIdx.x >> 5);
  if (e >= E) return;
  int lane = threadIdx.x & 31;
  int s = src[e], d = dst[e];
  float w = eexp[e] / denom[d];
  float4 v = ((const float4*)(h + (size_t)s * FDIM))[lane];
  float* od = out + (size_t)d * FDIM + lane * 4;
  atomicAdd(od + 0, v.x * w);
  atomicAdd(od + 1, v.y * w);
  atomicAdd(od + 2, v.z * w);
  atomicAdd(od + 3, v.w * w);
}

__global__ void k_agnn_finish(const float* __restrict__ h, const float* __restrict__ beta,
                              const float* __restrict__ amax, const float* __restrict__ denom,
                              float* __restrict__ out, int N) {
  int idx = blockIdx.x * blockDim.x + threadIdx.x;
  if (idx >= N * 32) return;
  int v = idx >> 5, f = (idx & 31) * 4;
  float ws = __expf(beta[0] - amax[v]) / denom[v];   // self-loop softmax weight
  size_t base = (size_t)v * FDIM + f;
#pragma unroll
  for (int j = 0; j < 4; ++j)
    out[base + j] = tanhf(out[base + j] + h[base + j] * ws);
}

// ---------------------------------------------------------------- pooling + LN + linear

__global__ __launch_bounds__(256) void k_pool(const float* __restrict__ h,
                                              const int* __restrict__ batch,
                                              float* __restrict__ sums,
                                              float* __restrict__ cnts, int N) {
  int v = blockIdx.x * 8 + (threadIdx.x >> 5);
  if (v >= N) return;
  int lane = threadIdx.x & 31;
  int g = batch[v];
  float4 x = ((const float4*)(h + (size_t)v * FDIM))[lane];
  float* sg = sums + (size_t)g * FDIM + lane * 4;
  atomicAdd(sg + 0, x.x);
  atomicAdd(sg + 1, x.y);
  atomicAdd(sg + 2, x.z);
  atomicAdd(sg + 3, x.w);
  if (lane == 0) atomicAdd(&cnts[g], 1.0f);
}

__global__ __launch_bounds__(256) void k_final(const float* __restrict__ sums,
                                               const float* __restrict__ cnts,
                                               const float* __restrict__ Wl,
                                               const float* __restrict__ bl,
                                               float* __restrict__ out, int G) {
  int g = blockIdx.x * 8 + (threadIdx.x >> 5);
  if (g >= G) return;
  int lane = threadIdx.x & 31;
  float cnt = fmaxf(cnts[g], 1.0f);
  float4 p = ((const float4*)(sums + (size_t)g * FDIM))[lane];
  p.x /= cnt; p.y /= cnt; p.z /= cnt; p.w /= cnt;
  float mu = waveReduceSum(p.x + p.y + p.z + p.w) * (1.0f / FDIM);
  float dx = p.x - mu, dy = p.y - mu, dz = p.z - mu, dw = p.w - mu;
  float var = waveReduceSum(dx * dx + dy * dy + dz * dz + dw * dw) * (1.0f / FDIM);
  float inv = rsqrtf(var + 1e-5f);
  float4 w = ((const float4*)Wl)[lane];
  float dot = waveReduceSum(dx * inv * w.x + dy * inv * w.y + dz * inv * w.z + dw * inv * w.w);
  if (lane == 0) out[g] = dot + bl[0];
}

// ---------------------------------------------------------------- host orchestration

static void run_agnn(const int* src, const int* dst, const float* h, float* out,
                     const float* beta, float* nrm, float* amax, float* denom,
                     float* alpha, float* eexp, int N, int E, hipStream_t stream) {
  dim3 blk(256);
  k_rownorm<<<(N + 7) / 8, blk, 0, stream>>>(h, nrm, N);
  k_fill_dev<<<(N + 255) / 256, blk, 0, stream>>>(amax, beta, N);   // self-loop alpha = beta
  k_alpha<<<(E + 7) / 8, blk, 0, stream>>>(src, dst, h, nrm, beta, alpha, amax, E);
  k_denom_init<<<(N + 255) / 256, blk, 0, stream>>>(beta, amax, denom, N);
  k_edge_exp<<<(E + 255) / 256, blk, 0, stream>>>(dst, alpha, amax, eexp, denom, E);
  hipMemsetAsync(out, 0, (size_t)N * FDIM * 4, stream);
  k_agnn_scatter<<<(E + 7) / 8, blk, 0, stream>>>(src, dst, eexp, denom, h, out, E);
  k_agnn_finish<<<((N * 32) + 255) / 256, blk, 0, stream>>>(h, beta, amax, denom, out, N);
}

extern "C" void kernel_launch(void* const* d_in, const int* in_sizes, int n_in,
                              void* d_out, int out_size, void* d_ws, size_t ws_size,
                              hipStream_t stream) {
  (void)n_in; (void)ws_size;
  const float* x     = (const float*)d_in[0];
  const int*   ei    = (const int*)d_in[1];
  const int*   batch = (const int*)d_in[2];
  const float* W[5]  = {(const float*)d_in[3], (const float*)d_in[5], (const float*)d_in[7],
                        (const float*)d_in[9], (const float*)d_in[11]};
  const float* B[5]  = {(const float*)d_in[4], (const float*)d_in[6], (const float*)d_in[8],
                        (const float*)d_in[10], (const float*)d_in[12]};
  const float* beta1 = (const float*)d_in[13];
  const float* beta2 = (const float*)d_in[14];
  const float* Wl    = (const float*)d_in[15];
  const float* bl    = (const float*)d_in[16];
  float* out = (float*)d_out;

  int N = in_sizes[0] / FDIM;
  int E = in_sizes[1] / 2;
  int G = out_size;
  const int* src = ei;        // edge_index[0]
  const int* dst = ei + E;    // edge_index[1]

  // carve workspace (both node-feature buffers ~51MB each -> L2 resident)
  char* p = (char*)d_ws;
  auto carve = [&](size_t bytes) -> char* {
    char* r = p;
    p += (bytes + 255) & ~(size_t)255;
    return r;
  };
  float*  bufA  = (float*)carve((size_t)N * FDIM * 4);
  float*  bufB  = (float*)carve((size_t)N * FDIM * 4);
  float*  deg   = (float*)carve((size_t)N * 4);
  float*  dis   = (float*)carve((size_t)N * 4);
  float*  nrm   = (float*)carve((size_t)N * 4);
  float*  amax  = (float*)carve((size_t)N * 4);
  float*  denom = (float*)carve((size_t)N * 4);
  float*  alpha = (float*)carve((size_t)E * 4);
  float*  eexp  = (float*)carve((size_t)E * 4);
  float*  sums  = (float*)carve((size_t)G * FDIM * 4);
  float*  cnts  = (float*)carve((size_t)G * 4);
  __bf16* Wpk   = (__bf16*)carve((size_t)8 * 4 * 32 * 16 * 2);   // 32 KB packed weights

  dim3 blk(256);

  // symmetric GCN normalization over A + I
  k_fill<<<(N + 255) / 256, blk, 0, stream>>>(deg, 1.0f, N);   // self-loop
  k_deg<<<(E + 255) / 256, blk, 0, stream>>>(dst, deg, E);
  k_rsqrt<<<(N + 255) / 256, blk, 0, stream>>>(deg, dis, N);

  // 5 GCN layers: pack W -> WMMA GEMM -> edge scatter -> self-loop + bias + tanh
  const float* hin = x;
  for (int l = 0; l < 5; ++l) {
    k_pack_w<<<64, blk, 0, stream>>>(W[l], Wpk);
    k_gemm_wmma<<<N / 32, blk, 0, stream>>>(hin, Wpk, bufB);
    hipMemsetAsync(bufA, 0, (size_t)N * FDIM * 4, stream);
    k_gcn_scatter<<<(E + 7) / 8, blk, 0, stream>>>(src, dst, dis, bufB, bufA, E);
    k_gcn_finish<<<((N * 32) + 255) / 256, blk, 0, stream>>>(bufB, dis, B[l], bufA, N);
    hin = bufA;
  }

  // 2 AGNN layers (ping-pong bufA <-> bufB)
  run_agnn(src, dst, bufA, bufB, beta1, nrm, amax, denom, alpha, eexp, N, E, stream);
  run_agnn(src, dst, bufB, bufA, beta2, nrm, amax, denom, alpha, eexp, N, E, stream);

  // global mean pool + layernorm + linear
  hipMemsetAsync(sums, 0, (size_t)G * FDIM * 4, stream);
  hipMemsetAsync(cnts, 0, (size_t)G * 4, stream);
  k_pool<<<(N + 7) / 8, blk, 0, stream>>>(bufA, batch, sums, cnts, N);
  k_final<<<(G + 7) / 8, blk, 0, stream>>>(sums, cnts, Wl, bl, out, G);
}
